// LSTM_52261162058555
// MI455X (gfx1250) — matrix operand order repaired and checked by reading, not compiled, not run on hardware
//
#include <hip/hip_runtime.h>

typedef __attribute__((ext_vector_type(16))) _Float16 v16h;
typedef __attribute__((ext_vector_type(8)))  _Float16 v8h;
typedef __attribute__((ext_vector_type(8)))  float    v8f;

#define B_   128
#define T_   1024
#define IN_  128
#define H_   1024
#define NG_  4096   // 4*H
#define NC_  28
#define GRID_ 128   // persistent blocks per LSTM layer

// ---------------- utility kernels ----------------
__global__ void k_f32_to_f16(const float* __restrict__ src, _Float16* __restrict__ dst, int n) {
  int i = blockIdx.x * blockDim.x + threadIdx.x;
  if (i < n) dst[i] = (_Float16)src[i];
}

__global__ void k_fill_f16(_Float16* __restrict__ p, int n) {
  int i = blockIdx.x * blockDim.x + threadIdx.x;
  if (i < n) p[i] = (_Float16)0.0f;
}

__global__ void k_fill_f32(float* __restrict__ p, int n) {
  int i = blockIdx.x * blockDim.x + threadIdx.x;
  if (i < n) p[i] = 0.0f;
}

// Pack LSTM weights: Wcat[n][k] (k contiguous), n = 4*j + g <-> original row r = g*H + j.
__global__ void k_pack_lstm_w(const float* __restrict__ Wih, const float* __restrict__ Whh,
                              const float* __restrict__ bih, const float* __restrict__ bhh,
                              _Float16* __restrict__ Wcat, float* __restrict__ bcat,
                              int Din, int Ktot) {
  size_t idx = (size_t)blockIdx.x * blockDim.x + threadIdx.x;
  size_t total = (size_t)NG_ * (size_t)Ktot;
  if (idx >= total) return;
  int n = (int)(idx / Ktot);
  int k = (int)(idx % Ktot);
  int j = n >> 2, g = n & 3;
  int r = g * H_ + j;
  float w = (k < Din) ? Wih[(size_t)r * Din + k] : Whh[(size_t)r * H_ + (k - Din)];
  Wcat[idx] = (_Float16)w;
  if (k == 0) bcat[n] = bih[r] + bhh[r];
}

__global__ void k_pack_fc_w(const float* __restrict__ W, _Float16* __restrict__ Wp,
                            int N, int Npad, int K) {
  size_t idx = (size_t)blockIdx.x * blockDim.x + threadIdx.x;
  if (idx >= (size_t)Npad * (size_t)K) return;
  int n = (int)(idx / K);
  Wp[idx] = (_Float16)((n < N) ? W[idx] : 0.0f);
}

// Fold bias + batchnorm into per-column affine: out = dot*alpha + beta
__global__ void k_prep_affine(const float* __restrict__ fcb,
                              const float* __restrict__ g, const float* __restrict__ b,
                              const float* __restrict__ m, const float* __restrict__ v,
                              float* __restrict__ alpha, float* __restrict__ beta,
                              int N, int Npad) {
  int n = blockIdx.x * blockDim.x + threadIdx.x;
  if (n >= Npad) return;
  if (n >= N) { alpha[n] = 0.0f; beta[n] = 0.0f; return; }
  float s  = g ? (g[n] * rsqrtf(v[n] + 1e-5f)) : 1.0f;
  float sh = g ? (b[n] - m[n] * s) : 0.0f;
  alpha[n] = s;
  beta[n]  = fcb[n] * s + sh;
}

// ---------------- persistent fused LSTM layer ----------------
// Grid: 128 blocks x 128 threads. Block = 64 batch rows x 64 gate-columns
// (16 complete hidden units; columns gate-interleaved n = 4j+g).
// Block's weight slice (64 cols x Ktot f16, <=257 KB) staged into LDS ONCE;
// all T steps feed WMMA B-operands from ds_load only. K-loop is 2x unrolled
// and software-pipelined: A (global) and B (LDS) fragments for chunk k+32
// are in flight while chunk k's 4 WMMAs issue.
__launch_bounds__(128)
__global__ void k_lstm_layer(const _Float16* __restrict__ Xseq, // [B, T, Din] f16
                             const _Float16* __restrict__ Wcat, // [NG][Ktot] f16
                             const float*    __restrict__ bcat, // [NG]
                             _Float16*       __restrict__ h0,   // [B, H] f16 (even-t in)
                             _Float16*       __restrict__ h1,   // [B, H] f16 (odd-t in)
                             float*          __restrict__ cbuf, // [B, H] f32
                             _Float16*       __restrict__ yout, // [B, T, H] f16 or null
                             unsigned*       __restrict__ ctr,  // grid barrier counter
                             int Din, int Ktot) {
  extern __shared__ char smraw[];
  const int bstride = Ktot + 8;                       // halves; keeps 16B alignment
  _Float16* Blds = (_Float16*)smraw;                  // [64][bstride]
  float* gates = (float*)(smraw + (size_t)64 * bstride * 2);  // [64][65]

  const int tid  = threadIdx.x;
  const int lane = tid & 31;
  const int wave = tid >> 5;                 // 0..3 -> M-block
  const int l15  = lane & 15;
  const int hi   = lane >> 4;
  const int n0   = (blockIdx.x >> 1) * 64;   // gate-column base
  const int rb   = (blockIdx.x & 1) * 64;    // batch-row base
  const int row  = rb + wave * 16 + l15;     // batch row for A fragment

  // ---- stage B slice into LDS once ----
  {
    int c = tid >> 1, hh = tid & 1;
    int Kh = Ktot >> 1;
    const _Float16* src = Wcat + (size_t)(n0 + c) * Ktot + hh * Kh;
    _Float16* dst = Blds + (size_t)c * bstride + hh * Kh;
    for (int k = 0; k < Kh; k += 8)
      *(v8h*)(dst + k) = *(const v8h*)(src + k);
  }
  __syncthreads();

  const _Float16* bb0 = Blds + (size_t)l15 * bstride + hi * 16;
  const _Float16* bb1 = bb0 + (size_t)16 * bstride;
  const _Float16* bb2 = bb0 + (size_t)32 * bstride;
  const _Float16* bb3 = bb0 + (size_t)48 * bstride;

  const _Float16* xrow = Xseq + ((size_t)row * T_) * Din;  // advanced by Din per step

  for (int t = 0; t < T_; ++t, xrow += Din) {
    const _Float16* hin = (t & 1) ? h1 : h0;
    _Float16* hout = (t & 1) ? h0 : h1;
    const _Float16* hrow = hin + (size_t)row * H_;

    v8f acc0 = {}, acc1 = {}, acc2 = {}, acc3 = {};
    union AF { v16h v; v8h h[2]; };
    AF a0, a1;
    v16h b00, b01, b02, b03, b10, b11, b12, b13;

    auto loadA = [&](int k, AF& a) {
      const _Float16* ar = (k < Din) ? (xrow + k) : (hrow + (k - Din));
      a.h[0] = *(const v8h*)(ar + hi * 16);
      a.h[1] = *(const v8h*)(ar + hi * 16 + 8);
    };
    auto loadB = [&](int k, v16h& b0, v16h& b1, v16h& b2, v16h& b3) {
      b0 = *(const v16h*)(bb0 + k);
      b1 = *(const v16h*)(bb1 + k);
      b2 = *(const v16h*)(bb2 + k);
      b3 = *(const v16h*)(bb3 + k);
    };

    loadA(0, a0);
    loadB(0, b00, b01, b02, b03);
    for (int k0 = 0; k0 < Ktot; k0 += 64) {
      // prefetch chunk k0+32 (always in range: k0 <= Ktot-64)
      loadA(k0 + 32, a1);
      loadB(k0 + 32, b10, b11, b12, b13);
      acc0 = __builtin_amdgcn_wmma_f32_16x16x32_f16(false, a0.v, false, b00, (short)0, acc0, false, false);
      acc1 = __builtin_amdgcn_wmma_f32_16x16x32_f16(false, a0.v, false, b01, (short)0, acc1, false, false);
      acc2 = __builtin_amdgcn_wmma_f32_16x16x32_f16(false, a0.v, false, b02, (short)0, acc2, false, false);
      acc3 = __builtin_amdgcn_wmma_f32_16x16x32_f16(false, a0.v, false, b03, (short)0, acc3, false, false);
      int k2 = k0 + 64;
      if (k2 < Ktot) {                       // prefetch chunk k0+64
        loadA(k2, a0);
        loadB(k2, b00, b01, b02, b03);
      }
      acc0 = __builtin_amdgcn_wmma_f32_16x16x32_f16(false, a1.v, false, b10, (short)0, acc0, false, false);
      acc1 = __builtin_amdgcn_wmma_f32_16x16x32_f16(false, a1.v, false, b11, (short)0, acc1, false, false);
      acc2 = __builtin_amdgcn_wmma_f32_16x16x32_f16(false, a1.v, false, b12, (short)0, acc2, false, false);
      acc3 = __builtin_amdgcn_wmma_f32_16x16x32_f16(false, a1.v, false, b13, (short)0, acc3, false, false);
    }

    // D layout: VGPR r, lanes 0-15 -> M=r, lanes 16-31 -> M=r+8; N = lane&15
    {
      const int mloc = wave * 16 + hi * 8;
#pragma unroll
      for (int r = 0; r < 8; ++r) {
        gates[(mloc + r) * 65 + l15 +  0] = acc0[r];
        gates[(mloc + r) * 65 + l15 + 16] = acc1[r];
        gates[(mloc + r) * 65 + l15 + 32] = acc2[r];
        gates[(mloc + r) * 65 + l15 + 48] = acc3[r];
      }
    }
    // prefetch next timestep's x row into WGP cache (covers barrier wait)
    if (t + 1 < T_) {
      __builtin_prefetch(xrow + Din + hi * (Din / 2), 0, 1);
    }
    __syncthreads();

    // pointwise cell update: 64 rows x 16 units = 1024 elems, 8 per thread
#pragma unroll
    for (int p = 0; p < 8; ++p) {
      int idx = tid + p * 128;
      int rl = idx >> 4;            // local row 0..63
      int u  = idx & 15;            // local hidden unit
      int rr = rb + rl;             // batch row
      int jn = n0 + 4 * u;
      float gi = gates[rl * 65 + 4 * u + 0] + bcat[jn + 0];
      float gf = gates[rl * 65 + 4 * u + 1] + bcat[jn + 1];
      float gg = gates[rl * 65 + 4 * u + 2] + bcat[jn + 2];
      float go = gates[rl * 65 + 4 * u + 3] + bcat[jn + 3];
      float i_ = 1.0f / (1.0f + __expf(-gi));
      float f_ = 1.0f / (1.0f + __expf(-gf));
      float g_ = tanhf(gg);
      float o_ = 1.0f / (1.0f + __expf(-go));
      int j = (n0 >> 2) + u;        // global hidden unit
      size_t ci = (size_t)rr * H_ + j;
      float cn = f_ * cbuf[ci] + i_ * g_;
      float hn = o_ * tanhf(cn);
      cbuf[ci] = cn;
      hout[ci] = (_Float16)hn;
      if (yout) yout[((size_t)rr * T_ + t) * H_ + j] = (_Float16)hn;
    }

    // grid-wide barrier (monotonic counter, release/acquire at agent scope)
    __threadfence();
    __syncthreads();
    if (tid == 0) {
      unsigned target = (unsigned)gridDim.x * (unsigned)(t + 1);
      __hip_atomic_fetch_add(ctr, 1u, __ATOMIC_RELEASE, __HIP_MEMORY_SCOPE_AGENT);
      while (__hip_atomic_load(ctr, __ATOMIC_ACQUIRE, __HIP_MEMORY_SCOPE_AGENT) < target)
        __builtin_amdgcn_s_sleep(1);
    }
    __syncthreads();
  }
}

// ---------------- generic FC: out = relu?(A@W * alpha + beta) ----------------
__launch_bounds__(256)
__global__ void k_fc(const _Float16* __restrict__ A,   // [128, K]
                     const _Float16* __restrict__ W,   // [Npad][K]
                     const float* __restrict__ alpha,  // [Npad]
                     const float* __restrict__ beta,   // [Npad]
                     _Float16* __restrict__ out_h,     // [128, Npad] or null
                     float* __restrict__ out_f,        // [128, Nout] or null
                     int K, int Npad, int Nout, int relu) {
  const int lane = threadIdx.x & 31;
  const int wave = threadIdx.x >> 5;
  const int n0   = blockIdx.x * 64;
  const int l15  = lane & 15;
  const int hi   = lane >> 4;
  int ntiles = (Npad - n0) / 16;
  if (ntiles > 4) ntiles = 4;
  const int row = wave * 16 + l15;

  v8f acc[4] = {v8f{}, v8f{}, v8f{}, v8f{}};
  const _Float16* arow = A + (size_t)row * K;
  const _Float16* bcol = W + (size_t)(n0 + l15) * K + hi * 16;

  for (int k0 = 0; k0 < K; k0 += 32) {
    union { v16h v; v8h h[2]; } a;
    a.h[0] = *(const v8h*)(arow + k0 + hi * 16);
    a.h[1] = *(const v8h*)(arow + k0 + hi * 16 + 8);
    for (int nt = 0; nt < ntiles; ++nt) {
      v16h b = *(const v16h*)(bcol + (size_t)(16 * nt) * K + k0);
      acc[nt] = __builtin_amdgcn_wmma_f32_16x16x32_f16(false, a.v, false, b, (short)0, acc[nt], false, false);
    }
  }

  const int mbase = wave * 16 + hi * 8;
  for (int nt = 0; nt < ntiles; ++nt) {
    int col = n0 + nt * 16 + l15;
    float al = alpha[col], be = beta[col];
#pragma unroll
    for (int r = 0; r < 8; ++r) {
      float v = acc[nt][r] * al + be;
      if (relu) v = v > 0.0f ? v : 0.0f;
      int m = mbase + r;
      if (out_h) out_h[(size_t)m * Npad + col] = (_Float16)v;
      if (out_f && col < Nout) out_f[(size_t)m * Nout + col] = v;
    }
  }
}

// ---------------- host orchestration ----------------
extern "C" void kernel_launch(void* const* d_in, const int* in_sizes, int n_in,
                              void* d_out, int out_size, void* d_ws, size_t ws_size,
                              hipStream_t stream) {
  (void)in_sizes; (void)n_in; (void)out_size; (void)ws_size;
  const float* x = (const float*)d_in[0];
  const float *Wih[3], *Whh[3], *bih[3], *bhh[3];
  for (int l = 0; l < 3; ++l) {
    Wih[l] = (const float*)d_in[1 + 4 * l];
    Whh[l] = (const float*)d_in[2 + 4 * l];
    bih[l] = (const float*)d_in[3 + 4 * l];
    bhh[l] = (const float*)d_in[4 + 4 * l];
  }
  const float* fc1_w = (const float*)d_in[13]; const float* fc1_b = (const float*)d_in[14];
  const float* bn1_g = (const float*)d_in[15]; const float* bn1_b = (const float*)d_in[16];
  const float* bn1_m = (const float*)d_in[17]; const float* bn1_v = (const float*)d_in[18];
  const float* fc2_w = (const float*)d_in[19]; const float* fc2_b = (const float*)d_in[20];
  const float* bn2_g = (const float*)d_in[21]; const float* bn2_b = (const float*)d_in[22];
  const float* bn2_m = (const float*)d_in[23]; const float* bn2_v = (const float*)d_in[24];
  const float* fc3_w = (const float*)d_in[25]; const float* fc3_b = (const float*)d_in[26];

  // workspace carve-out
  char* wp = (char*)d_ws;
  auto carve = [&](size_t bytes) -> char* {
    char* p = wp; wp += (bytes + 255) & ~(size_t)255; return p;
  };
  _Float16* xf16  = (_Float16*)carve((size_t)B_ * T_ * IN_ * 2);
  _Float16* y0    = (_Float16*)carve((size_t)B_ * T_ * H_ * 2);
  _Float16* y1    = (_Float16*)carve((size_t)B_ * T_ * H_ * 2);
  const int Kt[3] = {IN_ + H_, H_ + H_, H_ + H_};
  _Float16* Wcat[3]; float* bcat[3];
  for (int l = 0; l < 3; ++l) {
    Wcat[l] = (_Float16*)carve((size_t)NG_ * Kt[l] * 2);
    bcat[l] = (float*)carve((size_t)NG_ * 4);
  }
  _Float16* h_a = (_Float16*)carve((size_t)B_ * H_ * 2);
  _Float16* h_b = (_Float16*)carve((size_t)B_ * H_ * 2);
  float*    cb  = (float*)carve((size_t)B_ * H_ * 4);
  unsigned* ctr = (unsigned*)carve(256);
  _Float16* fc1wp = (_Float16*)carve((size_t)512 * 1024 * 2);
  _Float16* fc2wp = (_Float16*)carve((size_t)256 * 512 * 2);
  _Float16* fc3wp = (_Float16*)carve((size_t)32 * 256 * 2);
  float* al1 = (float*)carve(512 * 4); float* be1 = (float*)carve(512 * 4);
  float* al2 = (float*)carve(256 * 4); float* be2 = (float*)carve(256 * 4);
  float* al3 = (float*)carve(32 * 4);  float* be3 = (float*)carve(32 * 4);
  _Float16* z1 = (_Float16*)carve((size_t)B_ * 512 * 2);
  _Float16* z2 = (_Float16*)carve((size_t)B_ * 256 * 2);

  // allow large dynamic LDS for the persistent layer kernel (idempotent)
  const int Din[3] = {IN_, H_, H_};
  int smmax = (64 * (Kt[2] + 8) * 2) + (64 * 65 * 4);
  (void)hipFuncSetAttribute((const void*)k_lstm_layer,
                            hipFuncAttributeMaxDynamicSharedMemorySize, smmax);

  // ---- prep ----
  {
    int n = B_ * T_ * IN_;
    k_f32_to_f16<<<(n + 255) / 256, 256, 0, stream>>>(x, xf16, n);
  }
  for (int l = 0; l < 3; ++l) {
    size_t total = (size_t)NG_ * Kt[l];
    k_pack_lstm_w<<<(unsigned)((total + 255) / 256), 256, 0, stream>>>(
        Wih[l], Whh[l], bih[l], bhh[l], Wcat[l], bcat[l], Din[l], Kt[l]);
  }
  k_pack_fc_w<<<(512 * 1024 + 255) / 256, 256, 0, stream>>>(fc1_w, fc1wp, 512, 512, 1024);
  k_pack_fc_w<<<(256 * 512 + 255) / 256, 256, 0, stream>>>(fc2_w, fc2wp, 256, 256, 512);
  k_pack_fc_w<<<(32 * 256 + 255) / 256, 256, 0, stream>>>(fc3_w, fc3wp, NC_, 32, 256);
  k_prep_affine<<<2, 256, 0, stream>>>(fc1_b, bn1_g, bn1_b, bn1_m, bn1_v, al1, be1, 512, 512);
  k_prep_affine<<<1, 256, 0, stream>>>(fc2_b, bn2_g, bn2_b, bn2_m, bn2_v, al2, be2, 256, 256);
  k_prep_affine<<<1, 32, 0, stream>>>(fc3_b, nullptr, nullptr, nullptr, nullptr, al3, be3, NC_, 32);

  // ---- 3-layer LSTM (one persistent kernel per layer) ----
  const _Float16* Xseq[3] = {xf16, y0, y1};
  _Float16* Yout[3] = {y0, y1, nullptr};
  const int nh = B_ * H_;
  for (int l = 0; l < 3; ++l) {
    k_fill_f16<<<(nh + 255) / 256, 256, 0, stream>>>(h_a, nh);
    k_fill_f32<<<(nh + 255) / 256, 256, 0, stream>>>(cb, nh);
    k_fill_f32<<<1, 32, 0, stream>>>((float*)ctr, 1);
    int smbytes = (64 * (Kt[l] + 8) * 2) + (64 * 65 * 4);
    k_lstm_layer<<<GRID_, 128, smbytes, stream>>>(Xseq[l], Wcat[l], bcat[l],
                                                  h_a, h_b, cb, Yout[l], ctr,
                                                  Din[l], Kt[l]);
  }
  _Float16* hT = h_a;  // t = 1023 (odd) wrote h0 == h_a

  // ---- FC head ----
  k_fc<<<512 / 64, 256, 0, stream>>>(hT, fc1wp, al1, be1, z1, nullptr, 1024, 512, 512, 1);
  k_fc<<<256 / 64, 256, 0, stream>>>(z1, fc2wp, al2, be2, z2, nullptr, 512, 256, 256, 1);
  k_fc<<<1, 256, 0, stream>>>(z2, fc3wp, al3, be3, nullptr, (float*)d_out, 256, 32, NC_, 0);
}
